// SlotAttention_33956011442443
// MI455X (gfx1250) — matrix operand order repaired and checked by reading, not compiled, not run on hardware
//
#include <hip/hip_runtime.h>
#include <hip/hip_bf16.h>
#include <stdint.h>

typedef __bf16 bf16_t;
typedef __attribute__((ext_vector_type(16))) __bf16 v16bf;
typedef __attribute__((ext_vector_type(8)))  __bf16 v8bf;
typedef __attribute__((ext_vector_type(8)))  float  v8f;
typedef __attribute__((ext_vector_type(4)))  unsigned int v4u32;
typedef __attribute__((ext_vector_type(8)))  int v8i32;
typedef __attribute__((ext_vector_type(4)))  int v4i32;

#define WMMA_BF16(a, b, c) \
  __builtin_amdgcn_wmma_f32_16x16x32_bf16(false, (a), false, (b), (short)0, (c), false, false)

// A-fragment loader in contiguous k order (B frags use the same):
// lanes 0-15 hold K = {kc+0..15}; lanes 16-31 hold K = {kc+16..31}.
static __device__ __forceinline__ v16bf load_a_frag(const bf16_t* base, int kc, int hi) {
  union { v16bf v; v8bf h[2]; } u;
  u.h[0] = *(const v8bf*)(base + kc + hi * 16);
  u.h[1] = *(const v8bf*)(base + kc + hi * 16 + 8);
  return u.v;
}

// ---------------------------------------------------------------------------
// Generic GEMM: out(M,E) = A(M,K) * W(E,K)^T  (+bias, +relu, bf16/f32 out, +=)
// A, W in bf16; accumulate f32 via v_wmma_f32_16x16x32_bf16.
// Each wave: one E-tile (16 cols), four M-tiles (64 rows) -> B-frag reuse x4.
// grid.x = (M/64)*(E/16)/8, block = 256 (8 waves).
// ---------------------------------------------------------------------------
template<bool HAS_BIAS, bool RELU, bool OUT_BF16, bool ACC_ADD>
__global__ __launch_bounds__(256) void gemm_bf16_k(
    const bf16_t* __restrict__ A, const bf16_t* __restrict__ W,
    const float* __restrict__ bias, void* __restrict__ outp,
    int M, int K, int E)
{
  const int lane = threadIdx.x & 31;
  const int wave = threadIdx.x >> 5;
  const int le = lane & 15, hi = lane >> 4;
  const int etiles = E >> 4;
  const int gw = blockIdx.x * 8 + wave;
  const int mg = gw / etiles;
  const int et = gw - mg * etiles;
  const int m0 = mg * 64;

  const bf16_t* bptr = W + (size_t)(et * 16 + le) * K + hi * 16;
  const bf16_t* a0 = A + (size_t)(m0 + le) * K;
  const bf16_t* a1 = a0 + (size_t)16 * K;
  const bf16_t* a2 = a0 + (size_t)32 * K;
  const bf16_t* a3 = a0 + (size_t)48 * K;

  v8f acc[4] = {};
  for (int kc = 0; kc < K; kc += 32) {
    __builtin_prefetch(bptr + kc + 256, 0, 1);
    v16bf bf = *(const v16bf*)(bptr + kc);
    v16bf f0 = load_a_frag(a0, kc, hi);
    v16bf f1 = load_a_frag(a1, kc, hi);
    v16bf f2 = load_a_frag(a2, kc, hi);
    v16bf f3 = load_a_frag(a3, kc, hi);
    acc[0] = WMMA_BF16(f0, bf, acc[0]);
    acc[1] = WMMA_BF16(f1, bf, acc[1]);
    acc[2] = WMMA_BF16(f2, bf, acc[2]);
    acc[3] = WMMA_BF16(f3, bf, acc[3]);
  }

  float bv = 0.0f;
  if (HAS_BIAS) bv = bias[et * 16 + le];
  const int col = et * 16 + le;
  for (int t = 0; t < 4; ++t) {
    for (int r = 0; r < 8; ++r) {
      int row = m0 + t * 16 + r + hi * 8;   // C layout: VGPR r -> M=r (lanes 0-15), M=r+8 (16-31)
      size_t oi = (size_t)row * E + col;
      float v = acc[t][r] + bv;
      if (RELU) v = fmaxf(v, 0.0f);
      if (OUT_BF16)      ((bf16_t*)outp)[oi] = (bf16_t)v;
      else if (ACC_ADD)  ((float*)outp)[oi] += v;
      else               ((float*)outp)[oi] = v;
    }
  }
}

// ---------------------------------------------------------------------------
// LayerNorm over D=512, one wave per row, f32 in -> bf16 out.
// ---------------------------------------------------------------------------
__global__ __launch_bounds__(256) void ln_rows_k(
    const float* __restrict__ x, const float* __restrict__ g,
    const float* __restrict__ b, bf16_t* __restrict__ y, int rows)
{
  int row = blockIdx.x * 8 + (threadIdx.x >> 5);
  if (row >= rows) return;
  int lane = threadIdx.x & 31;
  const float* xr = x + (size_t)row * 512;
  float v[16];
  float s = 0.0f, ss = 0.0f;
  for (int i = 0; i < 16; ++i) {
    float t = xr[i * 32 + lane];
    v[i] = t; s += t; ss += t * t;
  }
  for (int m = 16; m >= 1; m >>= 1) {
    s  += __shfl_xor(s,  m, 32);
    ss += __shfl_xor(ss, m, 32);
  }
  float mean = s * (1.0f / 512.0f);
  float var  = ss * (1.0f / 512.0f) - mean * mean;
  float rstd = rsqrtf(var + 1e-5f);
  bf16_t* yr = y + (size_t)row * 512;
  for (int i = 0; i < 16; ++i) {
    int e = i * 32 + lane;
    yr[e] = (bf16_t)((v[i] - mean) * rstd * g[e] + b[e]);
  }
}

// ---------------------------------------------------------------------------
// Fused attention: q tile staged to LDS by the Tensor Data Mover (TDM),
// logits = k.q^T (WMMA), softmax over S=8 (+EPS renorm),
// partial updates = attn^T.v (WMMA, v transposed through LDS),
// atomicAdd into updates(B,S,D).
// grid = (N/256, B), block = 256 (8 waves); each block: 256 n-rows of one batch.
// ---------------------------------------------------------------------------
constexpr int ATS = 272;  // attnT row stride (bf16): 544 B/row, 32B-aligned
constexpr int VTS = 40;   // vT row stride (bf16): 80 B/row, 16B-aligned
constexpr int QS  = 528;  // q row stride (bf16): 1056 B/row = TDM pad (8 DW / 256 DW)

__global__ __launch_bounds__(256) void attn_update_k(
    const bf16_t* __restrict__ Km, const bf16_t* __restrict__ Vm,
    const bf16_t* __restrict__ Qm, float* __restrict__ updates)
{
  __shared__ bf16_t attnT[16 * ATS];   // [s][n]  (rows 8..15 stay zero)
  __shared__ bf16_t vT[512 * VTS];     // [d][n]  transposed v chunk (32 n per stage)
  __shared__ bf16_t qs[9 * QS];        // [s][d]  q tile (row 8 = zeros for pad slots)

  const int b = blockIdx.y;
  const int nbase = blockIdx.x * 256;
  const int lane = threadIdx.x & 31;
  const int wave = threadIdx.x >> 5;
  const int le = lane & 15, hi = lane >> 4;

  // --- TDM: DMA q[b] (8 rows x 512 bf16) into LDS, 8-DWORD pad per row ---
  if (wave == 0) {
    unsigned int ldsa = (unsigned int)(uintptr_t)(&qs[0]);
    unsigned long long ga = (unsigned long long)(uintptr_t)(Qm + (size_t)b * 8 * 512);
    v4u32 g0;
    g0.x = 1u;                                              // count=1 (valid user D#)
    g0.y = ldsa;                                            // lds_addr
    g0.z = (unsigned int)(ga & 0xFFFFFFFFu);                // global_addr[31:0]
    g0.w = (unsigned int)((ga >> 32) & 0x01FFFFFFu) | (2u << 30);  // [56:32] | type=2
    v8i32 g1;
    g1[0] = (1 << 16)     // data_size = 2 bytes
          | (1 << 20)     // pad_enable
          | (7 << 22)     // pad_interval: 256 DWORDs (= one 512-elem row)
          | (7 << 25);    // pad_amount: 8 DWORDs -> row stride 528 bf16
    g1[1] = (int)(512u << 16);          // tensor_dim0 = 512
    g1[2] = (int)(8u << 16);            // tensor_dim1 = 8
    g1[3] = (int)(512u << 16);          // tile_dim0 = 512
    g1[4] = 8;                          // tile_dim1 = 8 (tile_dim2 = 0)
    g1[5] = 512;                        // tensor_dim0_stride = 512
    g1[6] = 0;
    g1[7] = 0;
    v4i32 gz = {0, 0, 0, 0};
#if defined(__clang_major__) && (__clang_major__ >= 23)
    v8i32 gz8 = {0, 0, 0, 0, 0, 0, 0, 0};
    __builtin_amdgcn_tensor_load_to_lds(g0, g1, gz, gz, gz8, 0);
#else
    __builtin_amdgcn_tensor_load_to_lds(g0, g1, gz, gz, 0);
#endif
  }

  for (int i = threadIdx.x; i < 16 * ATS; i += 256) attnT[i] = (bf16_t)0.0f;
  for (int i = threadIdx.x; i < QS; i += 256) qs[8 * QS + i] = (bf16_t)0.0f;
  if (wave == 0) __builtin_amdgcn_s_wait_tensorcnt(0);
  __syncthreads();

  const bf16_t* kb = Km + ((size_t)b * 4096 + nbase) * 512;
  const int qrowi = (le < 8) ? le : 8;            // pad slots read the zero row
  const bf16_t* qrow = &qs[qrowi * QS + hi * 16];

  // ---- phase 1: logits + softmax, write attn^T into LDS ----
  for (int t = 0; t < 2; ++t) {
    int tile = wave * 2 + t;                         // 16-row n-tile, 0..15
    const bf16_t* arow = kb + (size_t)(tile * 16 + le) * 512;
    v8f acc = {};
    for (int kc = 0; kc < 512; kc += 32) {
      v16bf a = load_a_frag(arow, kc, hi);
      v16bf qb = *(const v16bf*)(qrow + kc);
      acc = WMMA_BF16(a, qb, acc);
    }
    // C layout: VGPR r -> (n = tile*16 + r + hi*8, slot = lane&7 in lanes 0-7 / 16-23)
    for (int r = 0; r < 8; ++r) {
      float lg = acc[r];
      float mx = lg;
      for (int m = 4; m >= 1; m >>= 1) mx = fmaxf(mx, __shfl_xor(mx, m, 8));
      float e = __expf(lg - mx);
      float se = e;
      for (int m = 4; m >= 1; m >>= 1) se += __shfl_xor(se, m, 8);
      float p = e / se + 1e-8f;                 // softmax + EPS
      float ps = p;
      for (int m = 4; m >= 1; m >>= 1) ps += __shfl_xor(ps, m, 8);
      float a = p / ps;                          // renormalize over slots
      if ((lane & 8) == 0) {                     // lanes 0-7 (row r), 16-23 (row r+8)
        int s = lane & 7;
        int nloc = tile * 16 + r + hi * 8;
        attnT[s * ATS + nloc] = (bf16_t)a;
      }
    }
  }
  __syncthreads();

  // ---- phase 2: updates(s,d) += attn^T(16x256) * v(256x512), K-chunks of 32 ----
  v8f uacc[4] = {};                              // 4 d-tiles per wave
  for (int nc = 0; nc < 256; nc += 32) {
    __syncthreads();                             // previous vT fully consumed
    const bf16_t* vb = Vm + ((size_t)b * 4096 + nbase + nc) * 512;
    for (int c = threadIdx.x; c < 2048; c += 256) {
      int row = c >> 6;                          // 0..31 (n within chunk)
      int dc = (c & 63) * 8;                     // 0..504
      uint4 dta = *(const uint4*)(vb + (size_t)row * 512 + dc);
      const bf16_t* d8 = (const bf16_t*)&dta;
      for (int j = 0; j < 8; ++j) vT[(dc + j) * VTS + row] = d8[j];
    }
    __syncthreads();
    for (int i = 0; i < 4; ++i) {
      int d0 = (wave * 4 + i) * 16;
      union { v16bf v; v8bf h[2]; } au;
      au.h[0] = *(const v8bf*)(&attnT[le * ATS + nc + hi * 8]);
      au.h[1] = *(const v8bf*)(&attnT[le * ATS + nc + 16 + hi * 8]);
      union { v16bf v; v8bf h[2]; } bu;
      const bf16_t* vrow = &vT[(d0 + le) * VTS + hi * 8];
      bu.h[0] = *(const v8bf*)(vrow);
      bu.h[1] = *(const v8bf*)(vrow + 16);
      uacc[i] = WMMA_BF16(au.v, bu.v, uacc[i]);
    }
  }

  if (hi == 0) {                                  // rows s=0..7 live in lanes 0-15
    for (int i = 0; i < 4; ++i) {
      int d0 = (wave * 4 + i) * 16;
      for (int r = 0; r < 8; ++r) {
        atomicAdd(&updates[((size_t)b * 8 + r) * 512 + d0 + le], uacc[i][r]);
      }
    }
  }
}

// ---------------------------------------------------------------------------
// Elementwise kernels
// ---------------------------------------------------------------------------
__global__ __launch_bounds__(256) void slots_init_k(
    const float* __restrict__ mu, const float* __restrict__ sg,
    const float* __restrict__ nz, float* __restrict__ slots, int n)
{
  int i = blockIdx.x * 256 + threadIdx.x;
  if (i < n) {
    int j = i & 4095;                             // (S*D)=4096
    slots[i] = mu[j] + sg[j] * nz[i];
  }
}

__global__ __launch_bounds__(256) void f2bf_k(
    const float* __restrict__ s, bf16_t* __restrict__ d, int n)
{
  int i = blockIdx.x * 256 + threadIdx.x;
  if (i < n) d[i] = (bf16_t)s[i];
}

__global__ __launch_bounds__(256) void gru_combine_k(
    const float* __restrict__ gi, const float* __restrict__ gh,
    float* __restrict__ slots, int n)
{
  int i = blockIdx.x * 256 + threadIdx.x;
  if (i >= n) return;
  int row = i >> 9, d = i & 511;
  const float* gir = gi + (size_t)row * 1536;
  const float* ghr = gh + (size_t)row * 1536;
  float r = 1.0f / (1.0f + __expf(-(gir[d] + ghr[d])));
  float z = 1.0f / (1.0f + __expf(-(gir[512 + d] + ghr[512 + d])));
  float nn = tanhf(gir[1024 + d] + r * ghr[1024 + d]);
  float prev = slots[i];
  slots[i] = (1.0f - z) * nn + z * prev;
}

// ---------------------------------------------------------------------------
// Host orchestration
// ---------------------------------------------------------------------------
extern "C" void kernel_launch(void* const* d_in, const int* in_sizes, int n_in,
                              void* d_out, int out_size, void* d_ws, size_t ws_size,
                              hipStream_t stream)
{
  (void)in_sizes; (void)n_in; (void)out_size; (void)ws_size;
  const float* inputs      = (const float*)d_in[0];
  const float* noise       = (const float*)d_in[1];
  const float* slots_mu    = (const float*)d_in[2];
  const float* slots_sigma = (const float*)d_in[3];
  const float* ln_in_g = (const float*)d_in[4];
  const float* ln_in_b = (const float*)d_in[5];
  const float* ln_s_g  = (const float*)d_in[6];
  const float* ln_s_b  = (const float*)d_in[7];
  const float* wq  = (const float*)d_in[8];
  const float* wk  = (const float*)d_in[9];
  const float* wv  = (const float*)d_in[10];
  const float* wih = (const float*)d_in[11];
  const float* whh = (const float*)d_in[12];
  const float* b_ih = (const float*)d_in[13];
  const float* b_hh = (const float*)d_in[14];
  const float* ln_m_g = (const float*)d_in[15];
  const float* ln_m_b = (const float*)d_in[16];
  const float* w1 = (const float*)d_in[17];
  const float* b1 = (const float*)d_in[18];
  const float* w2 = (const float*)d_in[19];
  const float* b2 = (const float*)d_in[20];

  float* slots = (float*)d_out;                   // (B,S,D) f32, recomputed every call

  char* ws = (char*)d_ws;
  size_t off = 0;
  auto alloc = [&](size_t bytes) -> char* {
    char* p = ws + off;
    off = (off + bytes + 255) & ~(size_t)255;
    return p;
  };
  const size_t BN = 131072;                       // B*N
  bf16_t* k_bf    = (bf16_t*)alloc(BN * 512 * 2);
  bf16_t* v_bf    = (bf16_t*)alloc(BN * 512 * 2);
  bf16_t* xn_bf   = (bf16_t*)alloc(BN * 512 * 2);
  bf16_t* q_bf    = (bf16_t*)alloc(256 * 512 * 2);
  bf16_t* sn_bf   = (bf16_t*)alloc(256 * 512 * 2);
  bf16_t* upd_bf  = (bf16_t*)alloc(256 * 512 * 2);
  bf16_t* slots_bf= (bf16_t*)alloc(256 * 512 * 2);
  bf16_t* hmid_bf = (bf16_t*)alloc(256 * 512 * 2);
  bf16_t* h_bf    = (bf16_t*)alloc(256 * 512 * 2);
  float*  updates = (float*)alloc(256 * 512 * 4);
  float*  gi      = (float*)alloc(256 * 1536 * 4);
  float*  gh      = (float*)alloc(256 * 1536 * 4);
  bf16_t* wq_bf   = (bf16_t*)alloc(512 * 512 * 2);
  bf16_t* wk_bf   = (bf16_t*)alloc(512 * 512 * 2);
  bf16_t* wv_bf   = (bf16_t*)alloc(512 * 512 * 2);
  bf16_t* w1_bf   = (bf16_t*)alloc(512 * 512 * 2);
  bf16_t* w2_bf   = (bf16_t*)alloc(512 * 512 * 2);
  bf16_t* wih_bf  = (bf16_t*)alloc(1536 * 512 * 2);
  bf16_t* whh_bf  = (bf16_t*)alloc(1536 * 512 * 2);

  auto gemm_grid = [](int M, int E) { return (M / 64) * (E / 16) / 8; };

  // slots = mu + sigma * noise
  slots_init_k<<<512, 256, 0, stream>>>(slots_mu, slots_sigma, noise, slots, 131072);

  // weights -> bf16 (once per call)
  f2bf_k<<<1024, 256, 0, stream>>>(wq,  wq_bf,  262144);
  f2bf_k<<<1024, 256, 0, stream>>>(wk,  wk_bf,  262144);
  f2bf_k<<<1024, 256, 0, stream>>>(wv,  wv_bf,  262144);
  f2bf_k<<<1024, 256, 0, stream>>>(w1,  w1_bf,  262144);
  f2bf_k<<<1024, 256, 0, stream>>>(w2,  w2_bf,  262144);
  f2bf_k<<<3072, 256, 0, stream>>>(wih, wih_bf, 786432);
  f2bf_k<<<3072, 256, 0, stream>>>(whh, whh_bf, 786432);

  // x = LN(inputs); k = x Wk^T; v = x Wv^T  (bf16, f32 accum)
  ln_rows_k<<<16384, 256, 0, stream>>>(inputs, ln_in_g, ln_in_b, xn_bf, 131072);
  gemm_bf16_k<false, false, true, false><<<gemm_grid(131072, 512), 256, 0, stream>>>(
      xn_bf, wk_bf, nullptr, k_bf, 131072, 512, 512);
  gemm_bf16_k<false, false, true, false><<<gemm_grid(131072, 512), 256, 0, stream>>>(
      xn_bf, wv_bf, nullptr, v_bf, 131072, 512, 512);

  for (int it = 0; it < 3; ++it) {
    // sn = LN(slots); q = sn Wq^T
    ln_rows_k<<<32, 256, 0, stream>>>(slots, ln_s_g, ln_s_b, sn_bf, 256);
    gemm_bf16_k<false, false, true, false><<<gemm_grid(256, 512), 256, 0, stream>>>(
        sn_bf, wq_bf, nullptr, q_bf, 256, 512, 512);

    // updates = attn^T v  (logits+softmax+renorm fused, q staged via TDM)
    hipMemsetAsync(updates, 0, 256 * 512 * 4, stream);
    attn_update_k<<<dim3(16, 32), 256, 0, stream>>>(k_bf, v_bf, q_bf, updates);

    // GRU: gi = updates Wih^T + b_ih ; gh = slots Whh^T + b_hh
    f2bf_k<<<512, 256, 0, stream>>>(updates, upd_bf, 131072);
    f2bf_k<<<512, 256, 0, stream>>>(slots, slots_bf, 131072);
    gemm_bf16_k<true, false, false, false><<<gemm_grid(256, 1536), 256, 0, stream>>>(
        upd_bf, wih_bf, b_ih, gi, 256, 512, 1536);
    gemm_bf16_k<true, false, false, false><<<gemm_grid(256, 1536), 256, 0, stream>>>(
        slots_bf, whh_bf, b_hh, gh, 256, 512, 1536);
    gru_combine_k<<<512, 256, 0, stream>>>(gi, gh, slots, 131072);

    // residual MLP: slots += relu(LN(slots) W1^T + b1) W2^T + b2
    ln_rows_k<<<32, 256, 0, stream>>>(slots, ln_m_g, ln_m_b, hmid_bf, 256);
    gemm_bf16_k<true, true, true, false><<<gemm_grid(256, 512), 256, 0, stream>>>(
        hmid_bf, w1_bf, b1, h_bf, 256, 512, 512);
    gemm_bf16_k<true, false, false, true><<<gemm_grid(256, 512), 256, 0, stream>>>(
        h_bf, w2_bf, b2, slots, 256, 512, 512);
  }
}
